// MultiHeadGSGAT_5738076307733
// MI455X (gfx1250) — matrix operand order repaired and checked by reading, not compiled, not run on hardware
//
#include <hip/hip_runtime.h>
#include <hip/hip_bf16.h>

// ---------------------------------------------------------------------------
// MultiHead GraphSAGE+GAT forward pass for MI455X (gfx1250), wave32 + WMMA.
// N=50000 nodes, E=500000 edges, IN=16, HD=128, EF=8, HEADS=4, NA=16, NG=64.
//
// Dense math -> v_wmma_f32_16x16x32_bf16 (fp32 accumulate), branch-free
// fragment builds (float2 loads + packed v_cvt_pk_bf16_f32 via convertvector).
// Irregular edge aggregation -> L2-resident float atomics (all node buffers
// fit in the 192 MB L2).
// ---------------------------------------------------------------------------

typedef __attribute__((ext_vector_type(16))) __bf16   v16bf;
typedef __attribute__((ext_vector_type(2)))  __bf16   bf16x2;
typedef __attribute__((ext_vector_type(2)))  float    f32x2;
typedef __attribute__((ext_vector_type(8)))  float    v8f;
typedef __attribute__((ext_vector_type(8)))  unsigned v8u;

#define IN_F    16
#define HD_F    128
#define EF_F    8
#define HEADS_F 4
#define NA_F    16
#define NG_F    64

#define GFLAG_RELU 1
#define GFLAG_ADDC 2

// ---- CDNA5 WMMA fragment coordinates (16x16x32 bf16) ----------------------
// A (16x32, MxK): lane -> M = lane%16, half = lane/16;
//   pair v (uint) holds K = half*16 + 2v and K+1 in this code (contiguous order).
// B (32x16, KxN): lane -> N = lane%16, half = lane/16;
//   pair v holds K = half*16 + 2v and K+1 in this code (same as A; not AMD's table order).
// C/D (16x16 f32): VGPR r, lane l -> M = r + 8*(l/16), N = l%16.

// packed f32x2 -> bf16x2 conversion (lowers to v_cvt_pk_bf16_f32)
__device__ inline unsigned pk_bf16(float lo, float hi) {
  f32x2 f; f[0] = lo; f[1] = hi;
  bf16x2 r = __builtin_convertvector(f, bf16x2);
  return __builtin_bit_cast(unsigned, r);
}
__device__ inline v16bf asbf(v8u u) { return __builtin_bit_cast(v16bf, u); }

// ===========================================================================
// Fill kernels
// ===========================================================================
__global__ void fill_f32(float* p, long long n, float v) {
  long long t = (long long)blockIdx.x * blockDim.x + threadIdx.x;
  if (t < n) p[t] = v;
}
__global__ void fill_u32(unsigned* p, long long n, unsigned v) {
  long long t = (long long)blockIdx.x * blockDim.x + threadIdx.x;
  if (t < n) p[t] = v;
}

// monotonic float<->uint encoding for atomicMax on floats
__device__ inline unsigned fenc(float f) {
  unsigned u = __float_as_uint(f);
  return (u & 0x80000000u) ? ~u : (u | 0x80000000u);
}
__device__ inline float fdec(unsigned u) {
  return (u & 0x80000000u) ? __uint_as_float(u & 0x7fffffffu)
                           : __uint_as_float(~u);
}

// ===========================================================================
// Fused NNConv edge kernel:
//   h1   = relu(edge_attr @ W1 + b1)          (16-edge tile, WMMA, K padded)
//   W_e  = h1 @ W2 + b2   (never materialized in HBM: saves ~1 GB traffic)
//   msg  = x[src] . W_e   -> atomicAdd into msg_sum[dst], deg[dst] += 1
// One wave per 16-edge tile, 8 waves / block.
// ===========================================================================
__launch_bounds__(256)
__global__ void nnconv_edge_kernel(const float* __restrict__ x,
                                   const int* __restrict__ srcI,
                                   const int* __restrict__ dstI,
                                   const float* __restrict__ ea,
                                   const float* __restrict__ W1,
                                   const float* __restrict__ b1,
                                   const float* __restrict__ W2,
                                   const float* __restrict__ b2,
                                   float* __restrict__ msg_sum,
                                   float* __restrict__ deg,
                                   int E)
{
  __shared__ unsigned W1p[16 * 32];    // W1 packed bf16 K-pairs, zero-padded
  __shared__ unsigned W2p[16 * 256];   // W2 packed bf16 K-pairs, 16 KB
  __shared__ float    b1s[32];
  __shared__ float    b2s[256];
  __shared__ float    h1s[8][16 * 32]; // per-wave h1 tile (fp32, post-relu)
  __shared__ float    xs[8][16 * 16];  // per-wave x[src] tile
  __shared__ int      dsts[8][16];

  const int tid  = threadIdx.x;
  const int wave = tid >> 5;
  const int lane = tid & 31;
  const int am   = lane & 15;
  const int half = lane >> 4;

  // stage packed weights: Wp[k2*ld + n] = pack(W[2k2][n], W[2k2+1][n])
  for (int i = tid; i < 16 * 256; i += 256) {
    int k2 = i >> 8, n = i & 255;
    W2p[i] = pk_bf16(W2[(2 * k2) * 256 + n], W2[(2 * k2 + 1) * 256 + n]);
  }
  for (int i = tid; i < 16 * 32; i += 256) {
    int k2 = i >> 5, n = i & 31;
    float f0 = (2 * k2     < EF_F) ? W1[(2 * k2) * 32 + n]     : 0.f;
    float f1 = (2 * k2 + 1 < EF_F) ? W1[(2 * k2 + 1) * 32 + n] : 0.f;
    W1p[i] = pk_bf16(f0, f1);
  }
  if (tid < 256) b2s[tid] = b2[tid];
  if (tid < 32)  b1s[tid] = b1[tid];
  __syncthreads();

  const int tile    = blockIdx.x * 8 + wave;
  const int base    = tile * 16;
  const bool active = (base < E);
  const int e  = base + am;
  const int ec = min(e, E - 1);

  // ---- A fragment: edge_attr 16x8, zero-padded to 16x32 ----
  // mask-by-multiply keeps the loads unconditional (no exec churn)
  v8u au = {};
  if (active) {
    const float msk = half ? 0.f : 1.f;
#pragma unroll
    for (int v = 0; v < 4; ++v) {       // K = 2v..2v+1 (valid for half==0)
      float2 p = *(const float2*)(ea + (long long)ec * EF_F + 2 * v);
      au[v] = pk_bf16(p.x * msk, p.y * msk);
    }
  }
  const v16bf aF = asbf(au);

  // ---- h1 = relu(ea @ W1 + b1), two N-tiles of 16 ----
#pragma unroll
  for (int nblk = 0; nblk < 2; ++nblk) {
    v8u bu;
#pragma unroll
    for (int v = 0; v < 8; ++v)         // zero-padding baked into W1p
      bu[v] = W1p[(half * 8 + v) * 32 + nblk * 16 + am];
    v8f c = {};
    c = __builtin_amdgcn_wmma_f32_16x16x32_bf16(false, aF, false, asbf(bu),
                                                (short)0, c, false, false);
    if (active) {
#pragma unroll
      for (int r = 0; r < 8; ++r) {
        int m = r + half * 8;
        h1s[wave][m * 32 + nblk * 16 + am] =
            fmaxf(c[r] + b1s[nblk * 16 + am], 0.f);
      }
    }
  }

  // stage x[src] tile and dst ids
  if (active) {
    int s = srcI[ec];
#pragma unroll
    for (int j = 0; j < 8; j += 2) {
      float2 p = *(const float2*)(x + (long long)s * IN_F + half * 8 + j);
      xs[wave][am * 16 + half * 8 + j]     = p.x;
      xs[wave][am * 16 + half * 8 + j + 1] = p.y;
    }
    if (half == 0) dsts[wave][am] = dstI[ec];
  }
  __syncthreads();

  // A fragment of h1 (reused across all 16 i-steps)
  v8u hu;
#pragma unroll
  for (int v = 0; v < 8; ++v) {
    int k = half * 16 + v * 2;
    float2 p = *(const float2*)&h1s[wave][am * 32 + k];
    hu[v] = pk_bf16(p.x, p.y);
  }
  const v16bf hF = asbf(hu);

  // msg[e,o] = sum_i x_src[e,i] * (h1[e,:] @ W2[:, i*16+o] + b2[i*16+o])
  v8f msg = {};
#pragma unroll 1
  for (int i = 0; i < 16; ++i) {
    v8u bu;
#pragma unroll
    for (int v = 0; v < 8; ++v)
      bu[v] = W2p[(half * 8 + v) * 256 + i * 16 + am];   // one ds_load_b32 each
    v8f c = {};
    c = __builtin_amdgcn_wmma_f32_16x16x32_bf16(false, hF, false, asbf(bu),
                                                (short)0, c, false, false);
#pragma unroll
    for (int r = 0; r < 8; ++r)
      msg[r] += xs[wave][(r + half * 8) * 16 + i] * (c[r] + b2s[i * 16 + am]);
  }

  if (active) {
#pragma unroll
    for (int r = 0; r < 8; ++r) {
      int m = r + half * 8;
      if (base + m < E)
        atomicAdd(&msg_sum[dsts[wave][m] * IN_F + am], msg[r]);
    }
    if (lane < 16 && base + lane < E)
      atomicAdd(&deg[dsts[wave][lane]], 1.0f);
  }
}

// ===========================================================================
// Branch-free WMMA GEMM: C[M,N] = (A/rowdiv) @ B [+C] + bias (+resid) (relu)
// REQUIRES: K % 32 == 0, N % 16 == 0 (grid.y = N/16), grid.x = ceil(M/16).
// A row-major [M,lda] (8B-aligned rows), B row-major [K,ldb].
// One wave per 16x16 output tile.
// ===========================================================================
__launch_bounds__(32)
__global__ void gemm_tile16(const float* __restrict__ A, int lda,
                            const float* __restrict__ B, int ldb,
                            const float* __restrict__ bias,
                            const float* __restrict__ rowdiv,
                            const float* __restrict__ resid,
                            float* __restrict__ C, int ldc,
                            int M, int K, int flags)
{
  const int lane = threadIdx.x;
  const int am   = lane & 15;
  const int half = lane >> 4;
  const int m0 = blockIdx.x * 16, n0 = blockIdx.y * 16;
  const int row = min(m0 + am, M - 1);          // clamp, no branch
  const int col = n0 + am;
  const float dsc = rowdiv ? 1.f / fmaxf(rowdiv[row], 1.f) : 1.f;

  v8f acc = {};
  for (int k0 = 0; k0 < K; k0 += 32) {
    v8u au, bu;
#pragma unroll
    for (int v = 0; v < 8; ++v) {
      int k = k0 + half * 16 + v * 2;
      float2 p = *(const float2*)(A + (long long)row * lda + k);
      au[v] = pk_bf16(p.x * dsc, p.y * dsc);
    }
#pragma unroll
    for (int v = 0; v < 8; ++v) {
      int k = k0 + half * 16 + v * 2;
      bu[v] = pk_bf16(B[(long long)k * ldb + col],
                      B[(long long)(k + 1) * ldb + col]);
    }
    acc = __builtin_amdgcn_wmma_f32_16x16x32_bf16(false, asbf(au), false,
                                                  asbf(bu), (short)0, acc,
                                                  false, false);
  }
  if (m0 + 16 <= M) {           // full tile: wave-uniform, unguarded stores
#pragma unroll
    for (int r = 0; r < 8; ++r) {
      int rr = m0 + r + half * 8;
      long long o = (long long)rr * ldc + col;
      float val = acc[r];
      if (flags & GFLAG_ADDC) val += C[o];
      if (bias)  val += bias[col];
      if (resid) val += resid[(long long)rr * ldc + col];
      if (flags & GFLAG_RELU) val = fmaxf(val, 0.f);
      C[o] = val;
    }
  } else {                      // ragged last tile
#pragma unroll
    for (int r = 0; r < 8; ++r) {
      int rr = m0 + r + half * 8;
      if (rr < M) {
        long long o = (long long)rr * ldc + col;
        float val = acc[r];
        if (flags & GFLAG_ADDC) val += C[o];
        if (bias)  val += bias[col];
        if (resid) val += resid[(long long)rr * ldc + col];
        if (flags & GFLAG_RELU) val = fmaxf(val, 0.f);
        C[o] = val;
      }
    }
  }
}

// ===========================================================================
// SAGE1 staging: cat[n, 0:16] = agg[n]/deg, cat[n, 16:32] = x0[n]
// ===========================================================================
__global__ void concat_mean_kernel(const float* __restrict__ agg,
                                   const float* __restrict__ deg,
                                   const float* __restrict__ x0,
                                   float* __restrict__ cat, int Nn)
{
  int t = blockIdx.x * blockDim.x + threadIdx.x;
  if (t >= Nn * 32) return;
  int n = t >> 5, o = t & 31;
  cat[t] = (o < 16) ? agg[n * 16 + o] / fmaxf(deg[n], 1.f)
                    : x0[n * 16 + (o - 16)];
}
__global__ void wcat_kernel(const float* __restrict__ Wl,
                            const float* __restrict__ Wr,
                            float* __restrict__ wcat)
{
  int t = blockIdx.x * blockDim.x + threadIdx.x;
  if (t >= 32 * HD_F) return;
  int k = t >> 7, n = t & 127;
  wcat[t] = (k < 16) ? Wl[k * HD_F + n] : Wr[(k - 16) * HD_F + n];
}

// ===========================================================================
// LayerNorm over D=128 (block of 128 threads, one row per block)
// ===========================================================================
__launch_bounds__(128)
__global__ void layernorm_kernel(float* __restrict__ X,
                                 const float* __restrict__ g,
                                 const float* __restrict__ b,
                                 int D, int relu_after)
{
  __shared__ float red[128];
  const int row = blockIdx.x;
  const int t = threadIdx.x;
  float v = X[(long long)row * D + t];
  red[t] = v; __syncthreads();
  for (int s = 64; s > 0; s >>= 1) { if (t < s) red[t] += red[t + s]; __syncthreads(); }
  float mean = red[0] / D; __syncthreads();
  float d = v - mean;
  red[t] = d * d; __syncthreads();
  for (int s = 64; s > 0; s >>= 1) { if (t < s) red[t] += red[t + s]; __syncthreads(); }
  float var = red[0] / D;
  float y = d * rsqrtf(var + 1e-5f) * g[t] + b[t];
  if (relu_after) y = fmaxf(y, 0.f);
  X[(long long)row * D + t] = y;
}

// ===========================================================================
// x0 = relu(msg_sum/deg + x @ nn_root + nn_bias)  (IN=16, small VALU kernel)
// ===========================================================================
__global__ void x0_kernel(const float* __restrict__ msg,
                          const float* __restrict__ deg,
                          const float* __restrict__ x,
                          const float* __restrict__ nn_root,
                          const float* __restrict__ nn_bias,
                          float* __restrict__ x0, int Nn)
{
  int t = blockIdx.x * blockDim.x + threadIdx.x;
  if (t >= Nn * IN_F) return;
  int n = t >> 4, o = t & 15;
  float acc = msg[t] / fmaxf(deg[n], 1.f) + nn_bias[o];
#pragma unroll
  for (int i = 0; i < IN_F; ++i) acc += x[n * IN_F + i] * nn_root[i * IN_F + o];
  x0[t] = fmaxf(acc, 0.f);
}

// ===========================================================================
// Scatter-add of gathered node features along edges: out[dst] += feat[src]
// ===========================================================================
__global__ void scatter_add_kernel(const float* __restrict__ feat,
                                   const int* __restrict__ srcI,
                                   const int* __restrict__ dstI,
                                   float* __restrict__ out,
                                   int E, int F)
{
  long long t = (long long)blockIdx.x * blockDim.x + threadIdx.x;
  if (t >= (long long)E * F) return;
  int e = (int)(t / F), o = (int)(t % F);
  atomicAdd(&out[(long long)dstI[e] * F + o], feat[(long long)srcI[e] * F + o]);
}

// ===========================================================================
// GAT per-node attention scores: ss[n,h] = <xh[n,h,:], a_s[h,:]>, sd likewise
// ===========================================================================
__global__ void gat_scores_kernel(const float* __restrict__ xh,
                                  const float* __restrict__ a_s,
                                  const float* __restrict__ a_d,
                                  float* __restrict__ ss,
                                  float* __restrict__ sd, int Nn)
{
  int t = blockIdx.x * blockDim.x + threadIdx.x;
  if (t >= Nn * HEADS_F) return;
  int n = t >> 2, h = t & 3;
  float s1 = 0.f, s2 = 0.f;
  for (int o = 0; o < HD_F; o += 2) {
    float2 p = *(const float2*)(xh + (long long)n * (HEADS_F * HD_F) + h * HD_F + o);
    s1 += p.x * a_s[h * HD_F + o] + p.y * a_s[h * HD_F + o + 1];
    s2 += p.x * a_d[h * HD_F + o] + p.y * a_d[h * HD_F + o + 1];
  }
  ss[t] = s1; sd[t] = s2;
}

__device__ inline float leaky02(float s) { return s > 0.f ? s : 0.2f * s; }

__global__ void gat_emax_kernel(const float* __restrict__ ss,
                                const float* __restrict__ sd,
                                const int* __restrict__ srcI,
                                const int* __restrict__ dstI,
                                unsigned* __restrict__ emax, int E)
{
  long long t = (long long)blockIdx.x * blockDim.x + threadIdx.x;
  if (t >= (long long)E * HEADS_F) return;
  int e = (int)(t >> 2), h = (int)(t & 3);
  float s = leaky02(ss[srcI[e] * HEADS_F + h] + sd[dstI[e] * HEADS_F + h]);
  atomicMax(&emax[dstI[e] * HEADS_F + h], fenc(s));
}

__global__ void gat_esum_kernel(const float* __restrict__ ss,
                                const float* __restrict__ sd,
                                const int* __restrict__ srcI,
                                const int* __restrict__ dstI,
                                const unsigned* __restrict__ emax,
                                float* __restrict__ esum, int E)
{
  long long t = (long long)blockIdx.x * blockDim.x + threadIdx.x;
  if (t >= (long long)E * HEADS_F) return;
  int e = (int)(t >> 2), h = (int)(t & 3);
  int d = dstI[e];
  float s = leaky02(ss[srcI[e] * HEADS_F + h] + sd[d * HEADS_F + h]);
  atomicAdd(&esum[d * HEADS_F + h], __expf(s - fdec(emax[d * HEADS_F + h])));
}

// out[dst,o] += 0.25 * sum_h alpha[e,h] * xh[src,h,o]   (block=128, grid=E)
__launch_bounds__(128)
__global__ void gat_aggregate_kernel(const float* __restrict__ xh,
                                     const float* __restrict__ ss,
                                     const float* __restrict__ sd,
                                     const unsigned* __restrict__ emax,
                                     const float* __restrict__ esum,
                                     const int* __restrict__ srcI,
                                     const int* __restrict__ dstI,
                                     float* __restrict__ out, int E)
{
  __shared__ float alpha[HEADS_F];
  int e = blockIdx.x;
  if (e >= E) return;
  int t = threadIdx.x;
  int s = srcI[e], d = dstI[e];
  if (t < HEADS_F) {
    float sc = leaky02(ss[s * HEADS_F + t] + sd[d * HEADS_F + t]);
    alpha[t] = __expf(sc - fdec(emax[d * HEADS_F + t])) / esum[d * HEADS_F + t];
  }
  __syncthreads();
  float acc = 0.f;
#pragma unroll
  for (int h = 0; h < HEADS_F; ++h)
    acc += alpha[h] * xh[(long long)s * (HEADS_F * HD_F) + h * HD_F + t];
  atomicAdd(&out[(long long)d * HD_F + t], 0.25f * acc);
}

// X = relu(X + bias (+resid)), element-wise
__global__ void bias_relu_kernel(float* __restrict__ X,
                                 const float* __restrict__ bias,
                                 const float* __restrict__ resid,
                                 long long total, int D)
{
  long long t = (long long)blockIdx.x * blockDim.x + threadIdx.x;
  if (t >= total) return;
  int o = (int)(t % D);
  float v = X[t] + bias[o];
  if (resid) v += resid[t];
  X[t] = fmaxf(v, 0.f);
}

// global mean pool accumulation
__global__ void pool_kernel(const float* __restrict__ x4,
                            const int* __restrict__ batch,
                            float* __restrict__ psum,
                            float* __restrict__ pcnt, int Nn)
{
  long long t = (long long)blockIdx.x * blockDim.x + threadIdx.x;
  if (t >= (long long)Nn * HD_F) return;
  int n = (int)(t >> 7), o = (int)(t & 127);
  int g = batch[n];
  atomicAdd(&psum[g * HD_F + o], x4[t]);
  if (o == 0) atomicAdd(&pcnt[g], 1.f);
}

// p_wire / p_term heads: out[t] = <x4[idx[t]], W> + b
__global__ void heads_kernel(const float* __restrict__ x4,
                             const int* __restrict__ wi,
                             const int* __restrict__ ti,
                             const float* __restrict__ wW, const float* __restrict__ wb,
                             const float* __restrict__ tW, const float* __restrict__ tb,
                             float* __restrict__ out)
{
  int t = blockIdx.x * blockDim.x + threadIdx.x;
  if (t >= 8192) return;
  const int*   idx = (t < 4096) ? wi : ti;
  const float* W   = (t < 4096) ? wW : tW;
  float        b   = (t < 4096) ? wb[0] : tb[0];
  int n = idx[t & 4095];
  float acc = b;
  for (int o = 0; o < HD_F; o += 2) {
    float2 p = *(const float2*)(x4 + (long long)n * HD_F + o);
    acc += p.x * W[o] + p.y * W[o + 1];
  }
  out[t] = acc;
}

// ===========================================================================
// Host-side orchestration
// ===========================================================================
static inline int cdiv(long long a, long long b) { return (int)((a + b - 1) / b); }

extern "C" void kernel_launch(void* const* d_in, const int* in_sizes, int n_in,
                              void* d_out, int out_size, void* d_ws, size_t ws_size,
                              hipStream_t stream) {
  // ---- inputs (setup_inputs dict order; params dict in insertion order) ----
  const float* x        = (const float*)d_in[0];
  const int*   wire_idx = (const int*)  d_in[1];
  const int*   term_idx = (const int*)  d_in[2];
  const int*   eidx     = (const int*)  d_in[3];
  const float* eattr    = (const float*)d_in[4];
  const int*   batch    = (const int*)  d_in[5];
  const float* enc_W1   = (const float*)d_in[6];
  const float* enc_b1   = (const float*)d_in[7];
  const float* enc_W2   = (const float*)d_in[8];
  const float* enc_b2   = (const float*)d_in[9];
  const float* nn_root  = (const float*)d_in[10];
  const float* nn_bias  = (const float*)d_in[11];
  const float* s1_Wl    = (const float*)d_in[12];
  const float* s1_bl    = (const float*)d_in[13];
  const float* s1_Wr    = (const float*)d_in[14];
  const float* s2_Wl    = (const float*)d_in[15];
  const float* s2_bl    = (const float*)d_in[16];
  const float* s2_Wr    = (const float*)d_in[17];
  const float* ln1_g    = (const float*)d_in[18];
  const float* ln1_b    = (const float*)d_in[19];
  const float* ln2_g    = (const float*)d_in[20];
  const float* ln2_b    = (const float*)d_in[21];
  const float* g1_W     = (const float*)d_in[22];
  const float* g1_as    = (const float*)d_in[23];
  const float* g1_ad    = (const float*)d_in[24];
  const float* g1_b     = (const float*)d_in[25];
  const float* g2_W     = (const float*)d_in[26];
  const float* g2_as    = (const float*)d_in[27];
  const float* g2_ad    = (const float*)d_in[28];
  const float* g2_b     = (const float*)d_in[29];
  const float* wire_W   = (const float*)d_in[30];
  const float* wire_b   = (const float*)d_in[31];
  const float* term_W   = (const float*)d_in[32];
  const float* term_b   = (const float*)d_in[33];
  const float* act_W1   = (const float*)d_in[34];
  const float* act_b1   = (const float*)d_in[35];
  const float* act_lng  = (const float*)d_in[36];
  const float* act_lnb  = (const float*)d_in[37];
  const float* act_W2   = (const float*)d_in[38];
  const float* act_b2   = (const float*)d_in[39];

  const int Nn = in_sizes[0] / IN_F;     // 50000
  const int E  = in_sizes[4] / EF_F;     // 500000
  const int* srcI = eidx;
  const int* dstI = eidx + E;
  float* out = (float*)d_out;

  // ---- workspace layout (floats) ----
  float* ws = (float*)d_ws;
  long long off = 0;
  float*    deg   = ws + off; off += Nn;                   // degrees (counts)
  float*    ss    = ws + off; off += (long long)Nn * 4;    // GAT src scores
  float*    sd    = ws + off; off += (long long)Nn * 4;    // GAT dst scores
  unsigned* emax  = (unsigned*)(ws + off); off += (long long)Nn * 4;
  float*    esum  = ws + off; off += (long long)Nn * 4;
  float*    t16   = ws + off; off += (long long)Nn * 16;   // msg_sum -> x0
  float*    cat   = ws + off; off += (long long)Nn * 32;   // sage1 [mean|x0]
  float*    bufA  = ws + off; off += (long long)Nn * 128;  // agg / gat2-out / x4
  float*    bufB  = ws + off; off += (long long)Nn * 128;  // x1 -> x3
  float*    bufC  = ws + off; off += (long long)Nn * 128;  // x2
  float*    xh    = ws + off; off += (long long)Nn * 512;  // GAT head features
  float*    psum  = ws + off; off += NG_F * HD_F;
  float*    pcnt  = ws + off; off += NG_F;
  float*    hA    = ws + off; off += NG_F * HD_F;
  float*    wcat  = ws + off; off += 32 * HD_F;            // sage1 [Wl;Wr]
  (void)ws_size; (void)n_in; (void)out_size;

  const int TB = 256;

  // ---- 1) NNConv: zero deg/msg, fused edge kernel, x0 finalize ----
  fill_f32<<<cdiv(Nn, TB), TB, 0, stream>>>(deg, Nn, 0.f);
  fill_f32<<<cdiv((long long)Nn * 16, TB), TB, 0, stream>>>(t16, (long long)Nn * 16, 0.f);
  nnconv_edge_kernel<<<cdiv(cdiv(E, 16), 8), 256, 0, stream>>>(
      x, srcI, dstI, eattr, enc_W1, enc_b1, enc_W2, enc_b2, t16, deg, E);
  x0_kernel<<<cdiv((long long)Nn * 16, TB), TB, 0, stream>>>(
      t16, deg, x, nn_root, nn_bias, t16, Nn);   // x0 overwrites msg_sum in place

  // ---- 2) SAGE1: agg(x0) -> bufA; x1 = LN(relu([mean|x0]@[Wl;Wr]+b)) -> bufB
  fill_f32<<<cdiv((long long)Nn * 16, TB), TB, 0, stream>>>(bufA, (long long)Nn * 16, 0.f);
  scatter_add_kernel<<<cdiv((long long)E * 16, TB), TB, 0, stream>>>(
      t16, srcI, dstI, bufA, E, 16);
  concat_mean_kernel<<<cdiv((long long)Nn * 32, TB), TB, 0, stream>>>(
      bufA, deg, t16, cat, Nn);
  wcat_kernel<<<cdiv(32 * HD_F, TB), TB, 0, stream>>>(s1_Wl, s1_Wr, wcat);
  {
    dim3 g(cdiv(Nn, 16), HD_F / 16);
    gemm_tile16<<<g, 32, 0, stream>>>(cat, 32, wcat, HD_F, s1_bl, nullptr,
                                      nullptr, bufB, HD_F, Nn, 32, GFLAG_RELU);
  }
  layernorm_kernel<<<Nn, 128, 0, stream>>>(bufB, ln1_g, ln1_b, HD_F, 0);

  // ---- 3) SAGE2: agg(x1) -> bufA; x2 = LN(relu(mean@Wl + x1@Wr + b + x1)) ----
  fill_f32<<<cdiv((long long)Nn * 128, TB), TB, 0, stream>>>(bufA, (long long)Nn * 128, 0.f);
  scatter_add_kernel<<<cdiv((long long)E * 128, TB), TB, 0, stream>>>(
      bufB, srcI, dstI, bufA, E, 128);
  {
    dim3 g(cdiv(Nn, 16), HD_F / 16);
    gemm_tile16<<<g, 32, 0, stream>>>(bufA, HD_F, s2_Wl, HD_F, nullptr, deg,
                                      nullptr, bufC, HD_F, Nn, HD_F, 0);
    gemm_tile16<<<g, 32, 0, stream>>>(bufB, HD_F, s2_Wr, HD_F, s2_bl, nullptr,
                                      bufB, bufC, HD_F, Nn, HD_F,
                                      GFLAG_RELU | GFLAG_ADDC);
  }
  layernorm_kernel<<<Nn, 128, 0, stream>>>(bufC, ln2_g, ln2_b, HD_F, 0);

  // ---- 4) GAT layers ----
  // GAT1: in=bufC(x2), out=bufB(x3, no resid). GAT2: in=bufB(x3), out=bufA(+x3).
  for (int layer = 0; layer < 2; ++layer) {
    const float* xin  = (layer == 0) ? bufC : bufB;
    float*       xout = (layer == 0) ? bufB : bufA;
    const float* W    = (layer == 0) ? g1_W  : g2_W;
    const float* a_s  = (layer == 0) ? g1_as : g2_as;
    const float* a_d  = (layer == 0) ? g1_ad : g2_ad;
    const float* bia  = (layer == 0) ? g1_b  : g2_b;
    const float* res  = (layer == 0) ? nullptr : bufB;   // + x3 for layer 2

    dim3 gx(cdiv(Nn, 16), (HEADS_F * HD_F) / 16);
    gemm_tile16<<<gx, 32, 0, stream>>>(xin, HD_F, W, HEADS_F * HD_F,
                                       nullptr, nullptr, nullptr,
                                       xh, HEADS_F * HD_F, Nn, HD_F, 0);
    gat_scores_kernel<<<cdiv((long long)Nn * 4, TB), TB, 0, stream>>>(
        xh, a_s, a_d, ss, sd, Nn);
    fill_u32<<<cdiv((long long)Nn * 4, TB), TB, 0, stream>>>(emax, (long long)Nn * 4, 0u);
    fill_f32<<<cdiv((long long)Nn * 4, TB), TB, 0, stream>>>(esum, (long long)Nn * 4, 0.f);
    fill_f32<<<cdiv((long long)Nn * 128, TB), TB, 0, stream>>>(xout, (long long)Nn * 128, 0.f);
    gat_emax_kernel<<<cdiv((long long)E * 4, TB), TB, 0, stream>>>(
        ss, sd, srcI, dstI, emax, E);
    gat_esum_kernel<<<cdiv((long long)E * 4, TB), TB, 0, stream>>>(
        ss, sd, srcI, dstI, emax, esum, E);
    gat_aggregate_kernel<<<E, 128, 0, stream>>>(
        xh, ss, sd, emax, esum, srcI, dstI, xout, E);
    bias_relu_kernel<<<cdiv((long long)Nn * 128, TB), TB, 0, stream>>>(
        xout, bia, res, (long long)Nn * 128, HD_F);
  }
  float* x4 = bufA;

  // ---- 5) node heads: p_wire (out[0:4096]) and p_term (out[4096:8192]) ----
  heads_kernel<<<cdiv(8192, TB), TB, 0, stream>>>(
      x4, wire_idx, term_idx, wire_W, wire_b, term_W, term_b, out);

  // ---- 6) global mean pool + action MLP -> out[8192:9216] ----
  fill_f32<<<cdiv(NG_F * HD_F + NG_F, TB), TB, 0, stream>>>(psum, NG_F * HD_F + NG_F, 0.f);
  pool_kernel<<<cdiv((long long)Nn * 128, TB), TB, 0, stream>>>(x4, batch, psum, pcnt, Nn);
  {
    dim3 g1(cdiv(NG_F, 16), HD_F / 16);
    gemm_tile16<<<g1, 32, 0, stream>>>(psum, HD_F, act_W1, HD_F, act_b1, pcnt,
                                       nullptr, hA, HD_F, NG_F, HD_F, 0);
    layernorm_kernel<<<NG_F, 128, 0, stream>>>(hA, act_lng, act_lnb, HD_F, 1);
    dim3 g2(cdiv(NG_F, 16), NA_F / 16);
    gemm_tile16<<<g2, 32, 0, stream>>>(hA, HD_F, act_W2, NA_F, act_b2, nullptr,
                                       nullptr, out + 8192, NA_F, NG_F, HD_F, 0);
  }
}
